// Head_19258633355334
// MI455X (gfx1250) — matrix operand order repaired and checked by reading, not compiled, not run on hardware
//
#include <hip/hip_runtime.h>
#include <hip/hip_bf16.h>

typedef __attribute__((ext_vector_type(16))) _Float16 v16h;
typedef __attribute__((ext_vector_type(8)))  _Float16 h8;
typedef __attribute__((ext_vector_type(8)))  float    v8f;
typedef __attribute__((ext_vector_type(4)))  float    f4;
typedef __attribute__((ext_vector_type(4)))  unsigned int v4u;
typedef __attribute__((ext_vector_type(8)))  int      v8i;
typedef __attribute__((ext_vector_type(4)))  int      v4i;

#define BDIM 8
#define TDIM 2048
#define CDIM 1024
#define HEADD 64

#if defined(__has_builtin)
# if __has_builtin(__builtin_amdgcn_tensor_load_to_lds) && \
     __has_builtin(__builtin_amdgcn_s_wait_tensorcnt)
#  define USE_TDM 1
# endif
# if __has_builtin(__builtin_amdgcn_s_wait_asynccnt)
#  define HAVE_WAIT_ASYNC 1
# endif
#endif

// ---------------------------------------------------------------------------
// Kernel 1: k = x @ W_k^T  (f32 in -> f16 out), WMMA f32_16x16x32_f16.
// Each wave computes a 16(row) x 64(head) tile, looping C in chunks of 32.
// HBM-stream bound (64 MB of x); W stays hot in L2.
// ---------------------------------------------------------------------------
__global__ __launch_bounds__(128) void proj_kernel(const float* __restrict__ x,
                                                   const float* __restrict__ Wk,
                                                   _Float16* __restrict__ kh) {
  const int lane = threadIdx.x & 31;
  const int wv   = threadIdx.x >> 5;
  const int tile = blockIdx.x * 4 + wv;      // 1024 tiles of 16 rows
  const int rowBase = tile * 16;
  const int l16 = lane & 15;
  const bool hi = (lane >= 16);

  v8f acc[4] = {};                           // 4 head tiles of 16x16 f32

  const float* xrow = x + (size_t)(rowBase + l16) * CDIM;
  const int aseg = hi ? 8 : 0;
  const int bseg = hi ? 8 : 0;

  for (int c0 = 0; c0 < CDIM; c0 += 32) {
    if (c0 + 128 < CDIM) __builtin_prefetch(xrow + c0 + 128, 0, 1);

    // A operand: 16-bit A 16x32 layout (two contiguous 8-elem K segments)
    v16h a;
    {
      f4 f0 = *(const f4*)(xrow + c0 + aseg);
      f4 f1 = *(const f4*)(xrow + c0 + aseg + 4);
      f4 f2 = *(const f4*)(xrow + c0 + 16 + aseg);
      f4 f3 = *(const f4*)(xrow + c0 + 16 + aseg + 4);
#pragma unroll
      for (int i = 0; i < 4; ++i) {
        a[i]      = (_Float16)f0[i];
        a[4 + i]  = (_Float16)f1[i];
        a[8 + i]  = (_Float16)f2[i];
        a[12 + i] = (_Float16)f3[i];
      }
    }

    // B operands: W_k^T chunk, 32(K=c) x 16(N=head); K contiguous per lane
#pragma unroll
    for (int n = 0; n < 4; ++n) {
      const float* wrow = Wk + (size_t)(n * 16 + l16) * CDIM + c0 + bseg;
      f4 g0 = ((const f4*)wrow)[0];
      f4 g1 = ((const f4*)wrow)[1];
      f4 g2 = ((const f4*)wrow)[4];
      f4 g3 = ((const f4*)wrow)[5];
      v16h b;
#pragma unroll
      for (int i = 0; i < 4; ++i) {
        b[i]      = (_Float16)g0[i];
        b[4 + i]  = (_Float16)g1[i];
        b[8 + i]  = (_Float16)g2[i];
        b[12 + i] = (_Float16)g3[i];
      }
      acc[n] = __builtin_amdgcn_wmma_f32_16x16x32_f16(
          false, a, false, b, (short)0, acc[n], false, false);
    }
  }

  // D layout: c[j] -> row = rowBase + j + (hi?8:0), head = n*16 + l16
#pragma unroll
  for (int n = 0; n < 4; ++n)
#pragma unroll
    for (int j = 0; j < 8; ++j) {
      int r = rowBase + j + (hi ? 8 : 0);
      kh[(size_t)r * HEADD + n * 16 + l16] = (_Float16)acc[n][j];
    }
}

// ---------------------------------------------------------------------------
// TDM issue helper: DMA a 32x64 f16 tile (row stride 64 elems) global -> LDS.
// Descriptor per CDNA5 ISA ch.8: group0 = {flags, lds_addr, global_addr, type},
// group1 = {data_size, tensor dims, tile dims, stride}. 2D => groups 2/3 zero.
// amdgpu-toolchain builtin form: (v4u, v8i, v4i, v4i, v8i, i32 cpol).
// ---------------------------------------------------------------------------
#if defined(USE_TDM)
__device__ __forceinline__ void tdm_load_chunk(const _Float16* src,
                                               unsigned ldsOff) {
  unsigned long long ga = (unsigned long long)(uintptr_t)src;
  v4u g0;
  g0[0] = 1u;                                        // count=1, user mode
  g0[1] = ldsOff;                                    // lds_addr (bytes)
  g0[2] = (unsigned)(ga & 0xFFFFFFFFu);              // global_addr[31:0]
  g0[3] = (unsigned)((ga >> 32) & 0x01FFFFFFu) | (2u << 30); // [56:32]|type=2
  v8i g1;
  g1[0] = (int)(1u << 16);        // workgroup_mask=0, data_size=1 (2 bytes)
  g1[1] = (int)(64u << 16);       // tensor_dim0 = 64  (bits 79:48)
  g1[2] = (int)(2048u << 16);     // tensor_dim1 = 2048 (bits 111:80)
  g1[3] = (int)(64u << 16);       // tile_dim0 = 64    (bits 127:112)
  g1[4] = 32;                     // tile_dim1 = 32, tile_dim2 = 0
  g1[5] = 64;                     // tensor_dim0_stride = 64 elems
  g1[6] = 0;
  g1[7] = 0;
  v4i z4 = {0, 0, 0, 0};
  v8i z8 = {0, 0, 0, 0, 0, 0, 0, 0};
  __builtin_amdgcn_tensor_load_to_lds(g0, g1, z4, z4, z8, 0);
}
#endif

// ---------------------------------------------------------------------------
// Kernel 2: causal flash attention, q=k=v (f16 operands, f32 softmax/accum).
// Block = 4 waves = 64 query rows; key chunks of 32 double-buffered in LDS,
// filled by TDM (tensor_load_to_lds) or async global->LDS loads.
// ---------------------------------------------------------------------------
__global__ __launch_bounds__(128) void attn_kernel(const _Float16* __restrict__ kh,
                                                   float* __restrict__ out) {
  __shared__ _Float16 Ktile[2][32 * HEADD];  // 2 x 4 KB double buffer
  __shared__ _Float16 Ptile[4][16 * 32];     // 4 KB, per-wave P staging

  const int tid  = threadIdx.x;
  const int lane = tid & 31;
  const int wv   = tid >> 5;
  const int l16  = lane & 15;
  const bool hi  = (lane >= 16);

  const int qb        = blockIdx.x * 64;     // global row base (b*T + t)
  const int batchBase = (qb / TDIM) * TDIM;
  const int tq        = qb - batchBase;      // within-batch query base
  const int myRowBase = qb + wv * 16;

  // Load Q (A operands for h-chunks 0..31 and 32..63)
  v16h aq0, aq1;
  {
    const _Float16* qrow = kh + (size_t)(myRowBase + l16) * HEADD;
    const int s = hi ? 8 : 0;
    h8 q0 = *(const h8*)(qrow + s);
    h8 q1 = *(const h8*)(qrow + 16 + s);
    h8 q2 = *(const h8*)(qrow + 32 + s);
    h8 q3 = *(const h8*)(qrow + 48 + s);
#pragma unroll
    for (int i = 0; i < 8; ++i) {
      aq0[i] = q0[i]; aq0[8 + i] = q1[i];
      aq1[i] = q2[i]; aq1[8 + i] = q3[i];
    }
  }

  float m[8], lsum[8];
  v8f o[4] = {};
#pragma unroll
  for (int j = 0; j < 8; ++j) { m[j] = -1e30f; lsum[j] = 0.0f; }

  const float scale = 0.03125f;              // C^-0.5 = 1/32
  const int myRowT = tq + wv * 16 + (hi ? 8 : 0);
  const int nChunks = (tq + 64) / 32;        // causal: keys 0 .. tq+63

  const unsigned ldsK0 = (unsigned)(uintptr_t)(&Ktile[0][0]);
  const unsigned ldsK1 = (unsigned)(uintptr_t)(&Ktile[1][0]);
  const _Float16* srcBase = kh + (size_t)batchBase * HEADD;

  // Issue chunk 0 fill into buffer 0
#if defined(USE_TDM)
  if (wv == 0) tdm_load_chunk(srcBase, ldsK0);
#else
  {
    const unsigned loff = (unsigned)(wv * 1024 + lane * 16);
    unsigned long long gb = (unsigned long long)(uintptr_t)srcBase;
    asm volatile("global_load_async_to_lds_b128 %0, %1, %2\n\t"
                 "global_load_async_to_lds_b128 %0, %1, %2 offset:512"
                 :: "v"(ldsK0 + loff), "v"(loff), "s"(gb) : "memory");
  }
#endif

  for (int kc = 0; kc < nChunks; ++kc) {
    const _Float16* Kcur = (kc & 1) ? &Ktile[1][0] : &Ktile[0][0];

    // Wait for this chunk's DMA, make it visible block-wide
#if defined(USE_TDM)
    if (wv == 0) __builtin_amdgcn_s_wait_tensorcnt(0);
#else
# if defined(HAVE_WAIT_ASYNC)
    __builtin_amdgcn_s_wait_asynccnt(0);
# else
    asm volatile("s_wait_asynccnt 0x0" ::: "memory");
# endif
#endif
    __syncthreads();

    // Kick off next chunk into the other buffer (overlaps with compute)
    if (kc + 1 < nChunks) {
      const unsigned nxtLds = (kc & 1) ? ldsK0 : ldsK1;
      const _Float16* nsrc = srcBase + (size_t)(kc + 1) * 32 * HEADD;
#if defined(USE_TDM)
      if (wv == 0) tdm_load_chunk(nsrc, nxtLds);
#else
      const unsigned loff = (unsigned)(wv * 1024 + lane * 16);
      unsigned long long gb = (unsigned long long)(uintptr_t)nsrc;
      asm volatile("global_load_async_to_lds_b128 %0, %1, %2\n\t"
                   "global_load_async_to_lds_b128 %0, %1, %2 offset:512"
                   :: "v"(nxtLds + loff), "v"(loff), "s"(gb) : "memory");
#endif
    }

    const int keyBase = kc * 32;

    // Scores: S = Q . K^T (two 16-key tiles, contraction over 64 heads)
    v8f s[2];
#pragma unroll
    for (int nt = 0; nt < 2; ++nt) {
      const _Float16* kr = Kcur + (nt * 16 + l16) * HEADD + (hi ? 8 : 0);
      h8 x0 = *(const h8*)(kr);
      h8 x1 = *(const h8*)(kr + 16);
      h8 x2 = *(const h8*)(kr + 32);
      h8 x3 = *(const h8*)(kr + 48);
      v16h b0, b1;
#pragma unroll
      for (int i = 0; i < 8; ++i) {
        b0[i] = x0[i]; b0[8 + i] = x1[i];
        b1[i] = x2[i]; b1[8 + i] = x3[i];
      }
      v8f cz = {};
      cz = __builtin_amdgcn_wmma_f32_16x16x32_f16(false, aq0, false, b0,
                                                  (short)0, cz, false, false);
      cz = __builtin_amdgcn_wmma_f32_16x16x32_f16(false, aq1, false, b1,
                                                  (short)0, cz, false, false);
      s[nt] = cz;
    }

    // Scale + causal mask
#pragma unroll
    for (int nt = 0; nt < 2; ++nt) {
      const int key = keyBase + nt * 16 + l16;
#pragma unroll
      for (int j = 0; j < 8; ++j) {
        float v = s[nt][j] * scale;
        s[nt][j] = (key <= myRowT + j) ? v : -1e30f;
      }
    }

    // Online softmax update
    float alpha[8];
#pragma unroll
    for (int j = 0; j < 8; ++j) {
      float t = fmaxf(s[0][j], s[1][j]);
#pragma unroll
      for (int d = 1; d < 16; d <<= 1) t = fmaxf(t, __shfl_xor(t, d, 16));
      float mn = fmaxf(m[j], t);
      alpha[j] = __expf(m[j] - mn);
      m[j] = mn;
    }
    v8f p0, p1;
#pragma unroll
    for (int j = 0; j < 8; ++j) {
      p0[j] = __expf(s[0][j] - m[j]);
      p1[j] = __expf(s[1][j] - m[j]);
      float rs = p0[j] + p1[j];
#pragma unroll
      for (int d = 1; d < 16; d <<= 1) rs += __shfl_xor(rs, d, 16);
      lsum[j] = lsum[j] * alpha[j] + rs;
    }
#pragma unroll
    for (int n = 0; n < 4; ++n)
#pragma unroll
      for (int j = 0; j < 8; ++j) o[n][j] *= alpha[j];

    // Re-lay P (C-layout -> A-layout) through per-wave LDS tile
    {
      _Float16* pt = &Ptile[wv][0];
      const int mb = hi ? 8 : 0;
#pragma unroll
      for (int j = 0; j < 8; ++j) {
        pt[(mb + j) * 32 + l16]      = (_Float16)p0[j];
        pt[(mb + j) * 32 + 16 + l16] = (_Float16)p1[j];
      }
    }
    asm volatile("" ::: "memory");   // same-wave DS ops are HW-ordered
    v16h ap;
    {
      const _Float16* pr = &Ptile[wv][0] + l16 * 32 + (hi ? 16 : 0);
      h8 a0 = *(const h8*)(pr);
      h8 a1 = *(const h8*)(pr + 8);
#pragma unroll
      for (int i = 0; i < 8; ++i) { ap[i] = a0[i]; ap[8 + i] = a1[i]; }
    }

    // O += P . V  (V = K chunk; B operand strided from LDS)
#pragma unroll
    for (int n = 0; n < 4; ++n) {
      v16h bv;
      const int h   = n * 16 + l16;
      const int kb2 = hi ? 16 : 0;
#pragma unroll
      for (int i = 0; i < 16; ++i) bv[i] = Kcur[(kb2 + i) * HEADD + h];
      o[n] = __builtin_amdgcn_wmma_f32_16x16x32_f16(
          false, ap, false, bv, (short)0, o[n], false, false);
    }
    // no trailing barrier: next iteration's wait+barrier fences the buffers
  }

  // Normalize and write f32 output
#pragma unroll
  for (int n = 0; n < 4; ++n)
#pragma unroll
    for (int j = 0; j < 8; ++j) {
      int r = myRowBase + j + (hi ? 8 : 0);
      out[(size_t)r * HEADD + n * 16 + l16] = o[n][j] / lsum[j];
    }
}

extern "C" void kernel_launch(void* const* d_in, const int* in_sizes, int n_in,
                              void* d_out, int out_size, void* d_ws, size_t ws_size,
                              hipStream_t stream) {
  const float* x  = (const float*)d_in[0];   // [8,2048,1024] f32
  const float* Wk = (const float*)d_in[1];   // [64,1024] f32
  float* out = (float*)d_out;                // [8,2048,64] f32
  _Float16* kh = (_Float16*)d_ws;            // 2 MB f16 k-projection

  // 1024 row-tiles of 16, 4 waves per block
  proj_kernel<<<256, 128, 0, stream>>>(x, Wk, kh);
  // 256 blocks x 64 query rows (4 waves of 16)
  attn_kernel<<<256, 128, 0, stream>>>(kh, out);
}
